// StyleAggregationModule_53326313947597
// MI455X (gfx1250) — matrix operand order repaired and checked by reading, not compiled, not run on hardware
//
#include <hip/hip_runtime.h>
#include <math.h>

// ---------------------------------------------------------------------------
// Types for CDNA5 WMMA (wave32, V_WMMA_F32_16X16X32_BF16)
// ---------------------------------------------------------------------------
typedef __bf16 bf16_t;
typedef bf16_t v16bf __attribute__((ext_vector_type(16)));
typedef bf16_t v8bf  __attribute__((ext_vector_type(8)));
typedef float  v8f   __attribute__((ext_vector_type(8)));

#define BATCH  8
#define CDIM   512
#define NHEADS 8
#define HWQ    256      // H*W queries per batch
#define KHW    1024     // K*H*W keys per batch
#define CM     4096     // C * NHEADS
#define LSTRIDE 40      // LDS panel row stride (bf16 elems): 80B, 16B-aligned,
                        // 20-bank shift per row -> conflict-free 16-lane reads

__device__ __forceinline__ unsigned short f2bf(float f) {
  union { float f; unsigned int u; } x; x.f = f;
  unsigned int u = x.u;
  u += 0x7fffu + ((u >> 16) & 1u);          // round-to-nearest-even
  return (unsigned short)(u >> 16);
}

__device__ __forceinline__ v16bf mk16(v8bf lo, v8bf hi) {
  return __builtin_shufflevector(lo, hi, 0, 1, 2, 3, 4, 5, 6, 7,
                                 8, 9, 10, 11, 12, 13, 14, 15);
}

// ---------------------------------------------------------------------------
// Block GEMM core: 256 threads (8 waves) compute a 128x128 f32 macro-tile of
// A[128 x K] x Bt[128 rows of K]^T, staging 128x32 panels of both operands in
// LDS per K-step. Waves tile 4(M) x 2(N); each wave owns 32x64 = 2x4 WMMA
// accumulators. Fragment builds follow the ISA 16-bit A/B wave32 layouts:
//   A: lane<16 row r K[0..7]+K[16..23]; lane>=16 row r K[8..15]+K[24..31]
//   B: lane = column within 16-wide tile; lane<16 K[0..7]+K[16..23]; lane>=16 K[8..15]+K[24..31]
// ---------------------------------------------------------------------------
template <int KSIZE>
__device__ __forceinline__ void gemm_block_128x128(
    const unsigned short* __restrict__ Ag, int lda,   // A at macro-row start
    const unsigned short* __restrict__ Bg, int ldb,   // Bt at macro-col start
    v8f acc[2][4]) {
  __shared__ unsigned short lsA[128 * LSTRIDE];
  __shared__ unsigned short lsB[128 * LSTRIDE];

  const int tid  = threadIdx.x;
  const int wid  = tid >> 5;
  const int lane = tid & 31;
  const int r    = lane & 15;
  const int half = lane >> 4;
  const int wm   = wid & 3;       // wave row  -> rows wm*32
  const int wn   = wid >> 2;      // wave col  -> cols wn*64

  // staging: thread t moves 32B: row = t>>1, 16-elem chunk = (t&1)*16
  const int srow   = tid >> 1;
  const int schunk = (tid & 1) * 16;
  const unsigned short* gA = Ag + (size_t)srow * lda + schunk;
  const unsigned short* gB = Bg + (size_t)srow * ldb + schunk;
  unsigned short* sA = lsA + srow * LSTRIDE + schunk;
  unsigned short* sB = lsB + srow * LSTRIDE + schunk;

  const unsigned short* lA = lsA + (wm * 32 + r) * LSTRIDE + half * 8;
  const unsigned short* lB = lsB + (wn * 64 + r) * LSTRIDE + half * 8;

  for (int kk = 0; kk < KSIZE; kk += 32) {
    // global -> regs (issued before the barrier so loads overlap compute)
    v8bf a0 = *(const v8bf*)(gA + kk);
    v8bf a1 = *(const v8bf*)(gA + kk + 8);
    v8bf b0 = *(const v8bf*)(gB + kk);
    v8bf b1 = *(const v8bf*)(gB + kk + 8);
    __syncthreads();                       // previous panel fully consumed
    *(v8bf*)(sA)     = a0;
    *(v8bf*)(sA + 8) = a1;
    *(v8bf*)(sB)     = b0;
    *(v8bf*)(sB + 8) = b1;
    __syncthreads();                       // panel visible to all waves

    v16bf af[2];
#pragma unroll
    for (int mt = 0; mt < 2; ++mt) {
      const unsigned short* p = lA + mt * 16 * LSTRIDE;
      af[mt] = mk16(*(const v8bf*)(p), *(const v8bf*)(p + 16));
    }
#pragma unroll
    for (int j = 0; j < 4; ++j) {
      const unsigned short* p = lB + j * 16 * LSTRIDE;
      v16bf b = mk16(*(const v8bf*)(p), *(const v8bf*)(p + 16));
      acc[0][j] = __builtin_amdgcn_wmma_f32_16x16x32_bf16(
          false, af[0], false, b, (short)0, acc[0][j], false, false);
      acc[1][j] = __builtin_amdgcn_wmma_f32_16x16x32_bf16(
          false, af[1], false, b, (short)0, acc[1][j], false, false);
    }
  }
}

// ---------------------------------------------------------------------------
// Transpose f32 [R][Cc] -> bf16 [Cc][R], one slab per blockIdx.z (LDS tiled).
// All dims here are multiples of 32.
// ---------------------------------------------------------------------------
__global__ void transpose_bf16_kernel(const float* __restrict__ in,
                                      unsigned short* __restrict__ out,
                                      int R, int Cc) {
  __shared__ float tile[32][33];
  const size_t slab = (size_t)blockIdx.z * R * Cc;
  const int c0 = blockIdx.x * 32;
  const int r0 = blockIdx.y * 32;
#pragma unroll
  for (int j = 0; j < 32; j += 8) {
    const int r = r0 + threadIdx.y + j;
    const int c = c0 + threadIdx.x;
    tile[threadIdx.y + j][threadIdx.x] = in[slab + (size_t)r * Cc + c];
  }
  __syncthreads();
#pragma unroll
  for (int j = 0; j < 32; j += 8) {
    const int oc  = c0 + threadIdx.y + j;   // output row (original column)
    const int orr = r0 + threadIdx.x;       // output col (original row)
    out[slab + (size_t)oc * R + orr] = f2bf(tile[threadIdx.x][threadIdx.y + j]);
  }
}

// ---------------------------------------------------------------------------
// Projection GEMM: A[rows][512] x Wt[4096][512]^T + bias, scatter into
// per-head layout. vmode=0 -> dst[b][m][row][c]; vmode=1 -> dst[b][m][c][row].
// ---------------------------------------------------------------------------
__global__ void proj_gemm_kernel(const unsigned short* __restrict__ A,
                                 const unsigned short* __restrict__ Bt,
                                 const float* __restrict__ bias,
                                 unsigned short* __restrict__ dst,
                                 int rows_per_batch, int vmode) {
  const int rb = blockIdx.y * 128;
  const int cb = blockIdx.x * 128;
  v8f acc[2][4] = {};
  gemm_block_128x128<CDIM>(A + (size_t)rb * CDIM, CDIM,
                           Bt + (size_t)cb * CDIM, CDIM, acc);

  const int lane = threadIdx.x & 31;
  const int wid  = threadIdx.x >> 5;
  const int r = lane & 15, half = lane >> 4;
  const int wm = wid & 3, wn = wid >> 2;
#pragma unroll
  for (int mt = 0; mt < 2; ++mt) {
#pragma unroll
    for (int j = 0; j < 4; ++j) {
      const int n = cb + wn * 64 + j * 16 + r;
      const int m = n & (NHEADS - 1);           // proj column (c m) layout
      const int c = n >> 3;
      const float bv = bias[n];
#pragma unroll
      for (int rr = 0; rr < 8; ++rr) {
        const int row = rb + wm * 32 + mt * 16 + rr + half * 8;
        const int b = row / rows_per_batch;
        const int q = row - b * rows_per_batch;
        const size_t idx = vmode
            ? ((((size_t)b * NHEADS + m) * CDIM + c) * rows_per_batch + q)
            : ((((size_t)b * NHEADS + m) * rows_per_batch + q) * CDIM + c);
        dst[idx] = f2bf(acc[mt][j][rr] + bv);
      }
    }
  }
}

// ---------------------------------------------------------------------------
// scores[bm][q][k] = (Q_bm . K_bm^T) * inv_scale   (per-head 256x1024x512)
// grid: (KHW/128, HWQ/128, B*M)
// ---------------------------------------------------------------------------
__global__ void scores_gemm_kernel(const unsigned short* __restrict__ qh,
                                   const unsigned short* __restrict__ kh,
                                   float* __restrict__ scores,
                                   float inv_scale) {
  const int bm = blockIdx.z;
  const int rb = blockIdx.y * 128;
  const int cb = blockIdx.x * 128;
  v8f acc[2][4] = {};
  gemm_block_128x128<CDIM>(qh + ((size_t)bm * HWQ + rb) * CDIM, CDIM,
                           kh + ((size_t)bm * KHW + cb) * CDIM, CDIM, acc);

  const int lane = threadIdx.x & 31;
  const int wid  = threadIdx.x >> 5;
  const int r = lane & 15, half = lane >> 4;
  const int wm = wid & 3, wn = wid >> 2;
  float* srow = scores + (size_t)bm * HWQ * KHW;
#pragma unroll
  for (int mt = 0; mt < 2; ++mt) {
#pragma unroll
    for (int j = 0; j < 4; ++j) {
      const int col = cb + wn * 64 + j * 16 + r;
#pragma unroll
      for (int rr = 0; rr < 8; ++rr) {
        const int row = rb + wm * 32 + mt * 16 + rr + half * 8;
        srow[(size_t)row * KHW + col] = acc[mt][j][rr] * inv_scale;
      }
    }
  }
}

// ---------------------------------------------------------------------------
// Row softmax over 1024 f32 -> bf16 attn. One wave per row, shfl reductions.
// ---------------------------------------------------------------------------
__global__ void softmax_kernel(const float* __restrict__ scores,
                               unsigned short* __restrict__ attn) {
  const int row  = blockIdx.x * 8 + (threadIdx.x >> 5);
  const int lane = threadIdx.x & 31;
  const float4* rp = (const float4*)(scores + (size_t)row * KHW);
  float4 v[8];
  float mx = -3.4e38f;
#pragma unroll
  for (int j = 0; j < 8; ++j) {
    v[j] = rp[lane + j * 32];
    mx = fmaxf(mx, fmaxf(fmaxf(v[j].x, v[j].y), fmaxf(v[j].z, v[j].w)));
  }
#pragma unroll
  for (int s = 16; s > 0; s >>= 1) mx = fmaxf(mx, __shfl_xor(mx, s, 32));
  float sum = 0.f;
#pragma unroll
  for (int j = 0; j < 8; ++j) {
    v[j].x = __expf(v[j].x - mx); v[j].y = __expf(v[j].y - mx);
    v[j].z = __expf(v[j].z - mx); v[j].w = __expf(v[j].w - mx);
    sum += v[j].x + v[j].y + v[j].z + v[j].w;
  }
#pragma unroll
  for (int s = 16; s > 0; s >>= 1) sum += __shfl_xor(sum, s, 32);
  const float inv = 1.0f / sum;
  ushort4* ap = (ushort4*)(attn + (size_t)row * KHW);
#pragma unroll
  for (int j = 0; j < 8; ++j) {
    ushort4 o;
    o.x = f2bf(v[j].x * inv); o.y = f2bf(v[j].y * inv);
    o.z = f2bf(v[j].z * inv); o.w = f2bf(v[j].w * inv);
    ap[lane + j * 32] = o;
  }
}

// ---------------------------------------------------------------------------
// out[b][q][m*512+c] = attn_bm . V_bm   (per-head 256x512x1024, vT is [c][k])
// grid: (CDIM/128, HWQ/128, B*M)
// ---------------------------------------------------------------------------
__global__ void attnv_gemm_kernel(const unsigned short* __restrict__ attn,
                                  const unsigned short* __restrict__ vT,
                                  unsigned short* __restrict__ outb) {
  const int bm = blockIdx.z;
  const int rb = blockIdx.y * 128;
  const int cb = blockIdx.x * 128;
  v8f acc[2][4] = {};
  gemm_block_128x128<KHW>(attn + ((size_t)bm * HWQ + rb) * KHW, KHW,
                          vT + ((size_t)bm * CDIM + cb) * KHW, KHW, acc);

  const int lane = threadIdx.x & 31;
  const int wid  = threadIdx.x >> 5;
  const int r = lane & 15, half = lane >> 4;
  const int wm = wid & 3, wn = wid >> 2;
  const int b = bm >> 3, m = bm & 7;
#pragma unroll
  for (int mt = 0; mt < 2; ++mt) {
#pragma unroll
    for (int j = 0; j < 4; ++j) {
      const int c = cb + wn * 64 + j * 16 + r;
#pragma unroll
      for (int rr = 0; rr < 8; ++rr) {
        const int q = rb + wm * 32 + mt * 16 + rr + half * 8;
        outb[(size_t)(b * HWQ + q) * CM + m * CDIM + c] = f2bf(acc[mt][j][rr]);
      }
    }
  }
}

// ---------------------------------------------------------------------------
// fc GEMM: res(2048x512) = outb(2048x4096) x fcW + fcb, scattered to NCHW.
// grid: (512/128, 2048/128)
// ---------------------------------------------------------------------------
__global__ void fc_gemm_kernel(const unsigned short* __restrict__ outb,
                               const unsigned short* __restrict__ fcWt,
                               const float* __restrict__ fcb,
                               float* __restrict__ out) {
  const int rb = blockIdx.y * 128;
  const int cb = blockIdx.x * 128;
  v8f acc[2][4] = {};
  gemm_block_128x128<CM>(outb + (size_t)rb * CM, CM,
                         fcWt + (size_t)cb * CM, CM, acc);

  const int lane = threadIdx.x & 31;
  const int wid  = threadIdx.x >> 5;
  const int r = lane & 15, half = lane >> 4;
  const int wm = wid & 3, wn = wid >> 2;
#pragma unroll
  for (int mt = 0; mt < 2; ++mt) {
#pragma unroll
    for (int j = 0; j < 4; ++j) {
      const int c = cb + wn * 64 + j * 16 + r;
      const float bv = fcb[c];
#pragma unroll
      for (int rr = 0; rr < 8; ++rr) {
        const int row = rb + wm * 32 + mt * 16 + rr + half * 8;
        const int b  = row >> 8;                 // / HWQ
        const int hw = row & 255;
        out[(size_t)b * (2 * CDIM * HWQ) + (size_t)c * HWQ + hw] =
            acc[mt][j][rr] + bv;
      }
    }
  }
}

// Concat second half: out[b][512+c][hw] = content[b][c][hw]  (float4 copy)
__global__ void concat_kernel(const float* __restrict__ content,
                              float* __restrict__ out) {
  const int i4  = blockIdx.x * blockDim.x + threadIdx.x;  // 262144 float4
  const int b   = i4 >> 15;                               // 32768 float4/batch
  const int rem = i4 & 32767;
  const float4* src = (const float4*)content;
  float4* dst = (float4*)(out + (size_t)b * (2 * CDIM * HWQ) + CDIM * HWQ);
  dst[rem] = src[i4];
}

// ---------------------------------------------------------------------------
extern "C" void kernel_launch(void* const* d_in, const int* in_sizes, int n_in,
                              void* d_out, int out_size, void* d_ws, size_t ws_size,
                              hipStream_t stream) {
  (void)in_sizes; (void)n_in; (void)out_size; (void)ws_size;
  const float* content = (const float*)d_in[0];
  const float* refs    = (const float*)d_in[1];
  const float* qW  = (const float*)d_in[2];
  const float* qb  = (const float*)d_in[3];
  const float* kW  = (const float*)d_in[4];
  const float* kb  = (const float*)d_in[5];
  const float* vW  = (const float*)d_in[6];
  const float* vb  = (const float*)d_in[7];
  const float* fcW = (const float*)d_in[8];
  const float* fcb = (const float*)d_in[9];
  float* out = (float*)d_out;

  char* ws = (char*)d_ws;
  size_t off = 0;
  auto alloc = [&](size_t bytes) -> void* {
    void* p = ws + off;
    off += (bytes + 255) & ~(size_t)255;
    return p;
  };
  unsigned short* cs   = (unsigned short*)alloc((size_t)BATCH * HWQ * CDIM * 2);
  unsigned short* rs   = (unsigned short*)alloc((size_t)BATCH * KHW * CDIM * 2);
  unsigned short* qWt  = (unsigned short*)alloc((size_t)CM * CDIM * 2);
  unsigned short* kWt  = (unsigned short*)alloc((size_t)CM * CDIM * 2);
  unsigned short* vWt  = (unsigned short*)alloc((size_t)CM * CDIM * 2);
  unsigned short* fcWt = (unsigned short*)alloc((size_t)CDIM * CM * 2);
  unsigned short* qh   = (unsigned short*)alloc((size_t)BATCH * NHEADS * HWQ * CDIM * 2);
  unsigned short* kh   = (unsigned short*)alloc((size_t)BATCH * NHEADS * KHW * CDIM * 2);
  unsigned short* vT   = (unsigned short*)alloc((size_t)BATCH * NHEADS * CDIM * KHW * 2);
  float*          sc   = (float*)alloc((size_t)BATCH * NHEADS * HWQ * KHW * 4);
  unsigned short* attn = (unsigned short*)alloc((size_t)BATCH * NHEADS * HWQ * KHW * 2);
  unsigned short* outb = (unsigned short*)alloc((size_t)BATCH * HWQ * CM * 2);

  const dim3 tb(32, 8, 1);
  // activations: per-(b[,k]) slab is f32 [512][256] -> bf16 [256][512]
  transpose_bf16_kernel<<<dim3(8, 16, 8),  tb, 0, stream>>>(content, cs, 512, 256);
  transpose_bf16_kernel<<<dim3(8, 16, 32), tb, 0, stream>>>(refs,    rs, 512, 256);
  // weights: [512][4096] -> [4096][512]; fcW: [4096][512] -> [512][4096]
  transpose_bf16_kernel<<<dim3(128, 16, 1), tb, 0, stream>>>(qW,  qWt,  512, 4096);
  transpose_bf16_kernel<<<dim3(128, 16, 1), tb, 0, stream>>>(kW,  kWt,  512, 4096);
  transpose_bf16_kernel<<<dim3(128, 16, 1), tb, 0, stream>>>(vW,  vWt,  512, 4096);
  transpose_bf16_kernel<<<dim3(16, 128, 1), tb, 0, stream>>>(fcW, fcWt, 4096, 512);

  // projections (macro 128x128): q 2048x4096, k/v 8192x4096, K=512
  proj_gemm_kernel<<<dim3(32, 16), 256, 0, stream>>>(cs, qWt, qb, qh, HWQ, 0);
  proj_gemm_kernel<<<dim3(32, 64), 256, 0, stream>>>(rs, kWt, kb, kh, KHW, 0);
  proj_gemm_kernel<<<dim3(32, 64), 256, 0, stream>>>(rs, vWt, vb, vT, KHW, 1);

  // faithful scale: C ** (M/2) = 512^4 = 2^36
  const float inv_scale = 1.0f / powf((float)CDIM, (float)NHEADS * 0.5f);
  scores_gemm_kernel<<<dim3(8, 2, BATCH * NHEADS), 256, 0, stream>>>(qh, kh, sc, inv_scale);
  softmax_kernel<<<16384 / 8, 256, 0, stream>>>(sc, attn);
  attnv_gemm_kernel<<<dim3(4, 2, BATCH * NHEADS), 256, 0, stream>>>(attn, vT, outb);
  fc_gemm_kernel<<<dim3(4, 16), 256, 0, stream>>>(outb, fcWt, fcb, out);
  concat_kernel<<<262144 / 256, 256, 0, stream>>>(content, out);
}
